// _GraphNetBlock_63488206569667
// MI455X (gfx1250) — matrix operand order repaired and checked by reading, not compiled, not run on hardware
//
#include <hip/hip_runtime.h>

typedef __bf16 v16bf __attribute__((ext_vector_type(16)));
typedef float  v8f   __attribute__((ext_vector_type(8)));
typedef unsigned int v4u __attribute__((ext_vector_type(4)));
typedef unsigned int v8u __attribute__((ext_vector_type(8)));

// d_ws layout (ushort units): bf16 weight fragments, packed as (kt*8+nt)*512 + lane*16 + half
#define OFF_W0E 0        // 12 kt * 8 nt * 512 = 49152   (384x128)
#define OFF_W1E 49152    //  4 kt * 8 nt * 512 = 16384   (128x128)
#define OFF_W0N 65536    //  8 kt * 8 nt * 512 = 32768   (256x128)
#define OFF_W1N 98304    //  4 kt * 8 nt * 512 = 16384   (128x128)
#define WTOTAL  114688

__device__ __forceinline__ unsigned short f2bf(float f) {
  unsigned u = __builtin_bit_cast(unsigned, f);
  u += 0x7FFFu + ((u >> 16) & 1u);           // round-to-nearest-even
  return (unsigned short)(u >> 16);
}

// Pack f32 weights (fan_in x 128, row-major) into bf16 WMMA B fragments.
// B layout (16-bit, 32x16 per fragment): lane l holds column n = l&15,
// halves h=0..15 hold k = kt*32 + (l>=16 ? 16 : 0) + h  (contiguous per lane).
__global__ void pack_weights_kernel(const float* __restrict__ w0e, const float* __restrict__ w1e,
                                    const float* __restrict__ w0n, const float* __restrict__ w1n,
                                    unsigned short* __restrict__ ws) {
  int t = blockIdx.x * 256 + threadIdx.x;
  if (t >= WTOTAL) return;
  const float* W; int base;
  if (t < OFF_W1E)      { W = w0e; base = OFF_W0E; }
  else if (t < OFF_W0N) { W = w1e; base = OFF_W1E; }
  else if (t < OFF_W1N) { W = w0n; base = OFF_W0N; }
  else                  { W = w1n; base = OFF_W1N; }
  int tl = t - base;
  int f  = tl >> 9;            // fragment = kt*8 + nt
  int j  = tl & 511;
  int l  = j >> 4, h = j & 15;
  int kt = f >> 3, nt = f & 7;
  int k  = kt * 32 + ((l >= 16) ? 16 : 0) + h;
  int n  = nt * 16 + (l & 15);
  ws[t] = f2bf(W[k * 128 + n]);
}

// One MLP layer worth of WMMAs: A (16 x KT*32 bf16) in LDS row-major with row
// stride RS (ushorts), B fragments pre-packed in global.  acc += A*B over all kt.
// A layout (16-bit 16x32): lane l holds row m=l&15; halves 0..15 -> k=kt*32+ko+0..15,
// ko = (l>=16)*16  (two adjacent 16B chunks, contiguous).
template <int KT, int RS>
__device__ __forceinline__ void mma_layer(const unsigned short* Abase,
                                          const unsigned short* __restrict__ wsB,
                                          int lane, v8f acc[8]) {
  const int l15 = lane & 15;
  const int ko  = (lane >> 4) * 16;
  for (int kt = 0; kt < KT; ++kt) {
    const unsigned short* rp = Abase + l15 * RS + kt * 32 + ko;
    v4u c0 = *(const v4u*)rp;
    v4u c1 = *(const v4u*)(rp + 8);
    v8u av; av.lo = c0; av.hi = c1;
    v16bf afrag = __builtin_bit_cast(v16bf, av);
    const unsigned short* wp = wsB + (size_t)(kt * 8) * 512 + lane * 16;
#pragma unroll
    for (int nt = 0; nt < 8; ++nt) {
      v8u bv = *(const v8u*)(wp + nt * 512);
      v16bf bfrag = __builtin_bit_cast(v16bf, bv);
      acc[nt] = __builtin_amdgcn_wmma_f32_16x16x32_bf16(
          false, afrag, false, bfrag, (short)0, acc[nt], false, false);
    }
  }
}

// ---------------- edge update ----------------
__global__ __launch_bounds__(128) void edge_kernel(
    const float* __restrict__ nf, const float* __restrict__ ef,
    const long long* __restrict__ ei, const unsigned short* __restrict__ ws,
    const float* __restrict__ b0, const float* __restrict__ b1,
    const float* __restrict__ g, const float* __restrict__ bet,
    float* __restrict__ outE, float* agg, long long E) {
  __shared__ unsigned short A[4][16 * 392];   // 16 rows x 384 bf16 (+8 pad) per wave
  __shared__ long long sIdx[4][32];           // [0..15]=src, [16..31]=dst
  const int wave = threadIdx.x >> 5;
  const int lane = threadIdx.x & 31;
  long long e0 = ((long long)blockIdx.x * 4 + wave) * 16;
  const bool valid = (e0 + 16) <= E;
  if (!valid) e0 = 0;

  sIdx[wave][lane] = (lane < 16) ? ei[e0 + lane] : ei[E + e0 + (lane - 16)];
  __syncthreads();

  // Gather [nf[src] | nf[dst] | ef] -> bf16 LDS rows
  for (int r = 0; r < 16; ++r) {
    const long long s = sIdx[wave][r];
    const long long d = sIdx[wave][16 + r];
    unsigned short* rp = &A[wave][r * 392];
#pragma unroll
    for (int kk = 0; kk < 4; ++kk) {
      const int c = lane + kk * 32;
      rp[c]       = f2bf(nf[s * 128 + c]);
      rp[128 + c] = f2bf(nf[d * 128 + c]);
      rp[256 + c] = f2bf(ef[(e0 + r) * 128 + c]);
    }
  }
  __syncthreads();

  const int l15 = lane & 15;
  const int hh  = lane >> 4;
  const v8f zero = {0.f, 0.f, 0.f, 0.f, 0.f, 0.f, 0.f, 0.f};
  v8f acc[8];
#pragma unroll
  for (int nt = 0; nt < 8; ++nt) acc[nt] = zero;

  mma_layer<12, 392>(&A[wave][0], ws + OFF_W0E, lane, acc);   // 384 -> 128

  // bias + relu, re-fragment h through LDS (cols 0..127)
#pragma unroll
  for (int nt = 0; nt < 8; ++nt) {
    const float bb = b0[nt * 16 + l15];
#pragma unroll
    for (int i = 0; i < 8; ++i) {
      float v = acc[nt][i] + bb;
      v = v > 0.f ? v : 0.f;
      A[wave][(i + hh * 8) * 392 + nt * 16 + l15] = f2bf(v);
    }
  }
  __syncthreads();

#pragma unroll
  for (int nt = 0; nt < 8; ++nt) acc[nt] = zero;
  mma_layer<4, 392>(&A[wave][0], ws + OFF_W1E, lane, acc);    // 128 -> 128

#pragma unroll
  for (int nt = 0; nt < 8; ++nt) {
    const float bb = b1[nt * 16 + l15];
#pragma unroll
    for (int i = 0; i < 8; ++i) acc[nt][i] += bb;
  }

  // LayerNorm over 128 cols: each row lives on 16 lanes (8 vals each)
  float mu[8], rstd[8];
#pragma unroll
  for (int i = 0; i < 8; ++i) {
    float s = 0.f, q = 0.f;
#pragma unroll
    for (int nt = 0; nt < 8; ++nt) { const float v = acc[nt][i]; s += v; q += v * v; }
#pragma unroll
    for (int m = 1; m <= 8; m <<= 1) { s += __shfl_xor(s, m, 32); q += __shfl_xor(q, m, 32); }
    const float mean = s * (1.f / 128.f);
    mu[i] = mean;
    rstd[i] = rsqrtf(q * (1.f / 128.f) - mean * mean + 1e-5f);
  }

  if (valid) {
#pragma unroll
    for (int nt = 0; nt < 8; ++nt) {
      const int n = nt * 16 + l15;
      const float gg = g[n], bb = bet[n];
#pragma unroll
      for (int i = 0; i < 8; ++i) {
        const int m = i + hh * 8;
        const float y = (acc[nt][i] - mu[i]) * rstd[i] * gg + bb;
        const long long row = e0 + m;
        outE[row * 128 + n] = y + ef[row * 128 + n];
        const long long dd = sIdx[wave][16 + m];
        unsafeAtomicAdd(&agg[dd * 128 + n], y);
      }
    }
  }
}

// ---------------- node update ----------------
__global__ __launch_bounds__(128) void node_kernel(
    const float* __restrict__ nf, const float* agg,
    const unsigned short* __restrict__ ws,
    const float* __restrict__ b0, const float* __restrict__ b1,
    const float* __restrict__ g, const float* __restrict__ bet,
    float* outN, int N) {
  __shared__ unsigned short A[4][16 * 264];   // 16 rows x 256 bf16 (+8 pad) per wave
  const int wave = threadIdx.x >> 5;
  const int lane = threadIdx.x & 31;
  long long r0 = ((long long)blockIdx.x * 4 + wave) * 16;
  const bool valid = (r0 + 16) <= (long long)N;
  if (!valid) r0 = 0;

  for (int r = 0; r < 16; ++r) {
    const long long row = r0 + r;
    unsigned short* rp = &A[wave][r * 264];
#pragma unroll
    for (int kk = 0; kk < 4; ++kk) {
      const int c = lane + kk * 32;
      rp[c]       = f2bf(nf[row * 128 + c]);
      rp[128 + c] = f2bf(agg[row * 128 + c]);
    }
  }
  __syncthreads();

  const int l15 = lane & 15;
  const int hh  = lane >> 4;
  const v8f zero = {0.f, 0.f, 0.f, 0.f, 0.f, 0.f, 0.f, 0.f};
  v8f acc[8];
#pragma unroll
  for (int nt = 0; nt < 8; ++nt) acc[nt] = zero;

  mma_layer<8, 264>(&A[wave][0], ws + OFF_W0N, lane, acc);    // 256 -> 128

#pragma unroll
  for (int nt = 0; nt < 8; ++nt) {
    const float bb = b0[nt * 16 + l15];
#pragma unroll
    for (int i = 0; i < 8; ++i) {
      float v = acc[nt][i] + bb;
      v = v > 0.f ? v : 0.f;
      A[wave][(i + hh * 8) * 264 + nt * 16 + l15] = f2bf(v);
    }
  }
  __syncthreads();

#pragma unroll
  for (int nt = 0; nt < 8; ++nt) acc[nt] = zero;
  mma_layer<4, 264>(&A[wave][0], ws + OFF_W1N, lane, acc);    // 128 -> 128

#pragma unroll
  for (int nt = 0; nt < 8; ++nt) {
    const float bb = b1[nt * 16 + l15];
#pragma unroll
    for (int i = 0; i < 8; ++i) acc[nt][i] += bb;
  }

  float mu[8], rstd[8];
#pragma unroll
  for (int i = 0; i < 8; ++i) {
    float s = 0.f, q = 0.f;
#pragma unroll
    for (int nt = 0; nt < 8; ++nt) { const float v = acc[nt][i]; s += v; q += v * v; }
#pragma unroll
    for (int m = 1; m <= 8; m <<= 1) { s += __shfl_xor(s, m, 32); q += __shfl_xor(q, m, 32); }
    const float mean = s * (1.f / 128.f);
    mu[i] = mean;
    rstd[i] = rsqrtf(q * (1.f / 128.f) - mean * mean + 1e-5f);
  }

  if (valid) {
#pragma unroll
    for (int nt = 0; nt < 8; ++nt) {
      const int n = nt * 16 + l15;
      const float gg = g[n], bb = bet[n];
#pragma unroll
      for (int i = 0; i < 8; ++i) {
        const long long row = r0 + i + hh * 8;
        const float y = (acc[nt][i] - mu[i]) * rstd[i] * gg + bb;
        outN[row * 128 + n] = y + nf[row * 128 + n];
      }
    }
  }
}

extern "C" void kernel_launch(void* const* d_in, const int* in_sizes, int n_in,
                              void* d_out, int out_size, void* d_ws, size_t ws_size,
                              hipStream_t stream) {
  const float* nf      = (const float*)d_in[0];
  const float* ef      = (const float*)d_in[1];
  const long long* ei  = (const long long*)d_in[2];
  const float* e_w0    = (const float*)d_in[3];
  const float* e_b0    = (const float*)d_in[4];
  const float* e_w1    = (const float*)d_in[5];
  const float* e_b1    = (const float*)d_in[6];
  const float* e_g     = (const float*)d_in[7];
  const float* e_be    = (const float*)d_in[8];
  const float* n_w0    = (const float*)d_in[9];
  const float* n_b0    = (const float*)d_in[10];
  const float* n_w1    = (const float*)d_in[11];
  const float* n_b1    = (const float*)d_in[12];
  const float* n_g     = (const float*)d_in[13];
  const float* n_be    = (const float*)d_in[14];

  const int N      = in_sizes[0] / 128;
  const long long E = in_sizes[1] / 128;

  unsigned short* ws = (unsigned short*)d_ws;
  float* outN = (float*)d_out;
  float* outE = outN + (size_t)N * 128;
  float* agg  = outN;   // aggregate aliases the node-output region (read-before-write per row)

  hipMemsetAsync(outN, 0, (size_t)N * 128 * sizeof(float), stream);
  pack_weights_kernel<<<(WTOTAL + 255) / 256, 256, 0, stream>>>(e_w0, e_w1, n_w0, n_w1, ws);

  const int eblocks = (int)((E + 63) / 64);
  edge_kernel<<<eblocks, 128, 0, stream>>>(nf, ef, ei, ws, e_b0, e_b1, e_g, e_be, outE, agg, E);

  const int nblocks = (N + 63) / 64;
  node_kernel<<<nblocks, 128, 0, stream>>>(nf, agg, ws, n_b0, n_b1, n_g, n_be, outN, N);
}
